// Memory_efficient_network_893353198156
// MI455X (gfx1250) — matrix operand order repaired and checked by reading, not compiled, not run on hardware
//
#include <hip/hip_runtime.h>
#include <hip/hip_fp16.h>

typedef __attribute__((ext_vector_type(16))) _Float16 v16h;
typedef __attribute__((ext_vector_type(8)))  _Float16 v8h;
typedef __attribute__((ext_vector_type(8)))  float    v8f;

#define LEAKY(x) ((x) > 0.0f ? (x) : 0.01f * (x))
#define NEG_INF (-1e30f)

// Problem constants: B=2, C110=4, C011=2, C001=1, P=16, G=128, E=96, N=128, CO=2

// ---------------------------------------------------------------------------
// Weight prep: 5 f16 weight matrices [128][128] (two are fused sums)
//   mat0 = W1[0] (A: t110)        mat1 = W1[3] (D: t011)
//   mat2 = W1[1]+W1[5] (fused BF) mat3 = W1[2] (C: maxg t110)
//   mat4 = W1[4]+W1[6] (fused EG)
// ---------------------------------------------------------------------------
__global__ __launch_bounds__(256) void k_wprep(const float* __restrict__ W1,
                                               _Float16* __restrict__ wf) {
  int idx = blockIdx.x * 256 + threadIdx.x;
  if (idx >= 5 * 16384) return;
  int mat = idx / 16384, r = idx % 16384;
  float v;
  switch (mat) {
    case 0:  v = W1[0 * 16384 + r]; break;
    case 1:  v = W1[3 * 16384 + r]; break;
    case 2:  v = W1[1 * 16384 + r] + W1[5 * 16384 + r]; break;
    case 3:  v = W1[2 * 16384 + r]; break;
    default: v = W1[4 * 16384 + r] + W1[6 * 16384 + r]; break;
  }
  wf[idx] = (_Float16)v;
}

// ---------------------------------------------------------------------------
// Layer 0, stage A: per-b reduced pointwise terms (tiny channel counts)
//   GG0[b][n][g] = B0+F0 (g-only terms), EE0[b][n][e] = E0+G0, CC0[b][n][p] = C0
// ---------------------------------------------------------------------------
__global__ __launch_bounds__(128) void k_l0a(
    const float* __restrict__ x110, const float* __restrict__ x011,
    const float* __restrict__ x001,
    const float* __restrict__ W0_110, const float* __restrict__ b0_110,
    const float* __restrict__ W0_011, const float* __restrict__ b0_011,
    const float* __restrict__ W0_001, const float* __restrict__ b0_001,
    float* __restrict__ GG0, float* __restrict__ EE0, float* __restrict__ CC0) {
  int b = blockIdx.x;
  int t = threadIdx.x;
  __shared__ float mp110[4][128];   // max over p of x110[b,c,:,g]
  __shared__ float me011[2][128];   // max over e of x011[b,c,g,:]
  __shared__ float mg011[2][96];    // max over g of x011[b,c,:,e]
  __shared__ float mgx110[4][16];   // max over g of x110[b,c,p,:]
  __shared__ float x1r[96];

  for (int c = 0; c < 4; c++) {
    float m = NEG_INF;
    for (int p = 0; p < 16; p++) m = fmaxf(m, x110[((b * 4 + c) * 16 + p) * 128 + t]);
    mp110[c][t] = m;
  }
  for (int c = 0; c < 2; c++) {
    float m = NEG_INF;
    const float* xr = x011 + (b * 2 + c) * 12288 + t * 96;
    for (int e = 0; e < 96; e++) m = fmaxf(m, xr[e]);
    me011[c][t] = m;
  }
  if (t < 96) {
    for (int c = 0; c < 2; c++) {
      float m = NEG_INF;
      for (int g = 0; g < 128; g++) m = fmaxf(m, x011[(b * 2 + c) * 12288 + g * 96 + t]);
      mg011[c][t] = m;
    }
    x1r[t] = x001[b * 96 + t];
  }
  if (t < 16) {
    for (int c = 0; c < 4; c++) {
      float m = NEG_INF;
      for (int g = 0; g < 128; g++) m = fmaxf(m, x110[((b * 4 + c) * 16 + t) * 128 + g]);
      mgx110[c][t] = m;
    }
  }
  __syncthreads();

  int n = t;
  float w1a[4], w2a[4], wf2[2], wf1[2];
  for (int c = 0; c < 4; c++) { w1a[c] = W0_110[(128 + n) * 4 + c]; w2a[c] = W0_110[(256 + n) * 4 + c]; }
  for (int c = 0; c < 2; c++) { wf2[c] = W0_011[(256 + n) * 2 + c]; wf1[c] = W0_011[(128 + n) * 2 + c]; }
  float bB = b0_110[128 + n], bC = b0_110[256 + n];
  float bF = b0_011[256 + n], bE = b0_011[128 + n];
  float w001 = W0_001[n], b001v = b0_001[n];

  for (int g = 0; g < 128; g++) {
    float s = bB + bF;
    for (int c = 0; c < 4; c++) s += w1a[c] * mp110[c][g];
    for (int c = 0; c < 2; c++) s += wf2[c] * me011[c][g];
    GG0[(b * 128 + n) * 128 + g] = s;
  }
  for (int e = 0; e < 96; e++) {
    float s = bE + b001v + w001 * x1r[e];
    for (int c = 0; c < 2; c++) s += wf1[c] * mg011[c][e];
    EE0[(b * 128 + n) * 96 + e] = s;
  }
  for (int p = 0; p < 16; p++) {
    float s = bC;
    for (int c = 0; c < 4; c++) s += w2a[c] * mgx110[c][p];
    CC0[(b * 128 + n) * 16 + p] = s;
  }
}

// ---------------------------------------------------------------------------
// Layer 0, stage B: one block per (b,n). Produces f16 layer-1 GEMM inputs
// in [k][col] layout (coalesced writes):
//   t110h[k][b*2048+p*128+g], t011h[k][b*12288+e*128+g],
//   bfh[k][b*128+g] (= maxp t110 = maxe t011), mg110h[k][b*16+p], s001h[k][b*96+e]
// ---------------------------------------------------------------------------
__global__ __launch_bounds__(128) void k_l0b(
    const float* __restrict__ x110, const float* __restrict__ x011,
    const float* __restrict__ W0_110, const float* __restrict__ b0_110,
    const float* __restrict__ W0_011, const float* __restrict__ b0_011,
    const float* __restrict__ GG0, const float* __restrict__ EE0,
    const float* __restrict__ CC0,
    _Float16* __restrict__ t110h, _Float16* __restrict__ t011h,
    _Float16* __restrict__ bfh, _Float16* __restrict__ mg110h,
    _Float16* __restrict__ s001h) {
  int b = blockIdx.x >> 7, n = blockIdx.x & 127;
  int g = threadIdx.x;
  __shared__ float cc[16], ee[96], tl[16 * 128];

  if (g < 16) cc[g] = CC0[(b * 128 + n) * 16 + g];
  if (g < 96) ee[g] = EE0[(b * 128 + n) * 96 + g];
  float gg = GG0[(b * 128 + n) * 128 + g];
  float wA[4];
  for (int c = 0; c < 4; c++) wA[c] = W0_110[n * 4 + c];
  float bA = b0_110[n];
  float wD0 = W0_011[n * 2 + 0], wD1 = W0_011[n * 2 + 1], bD = b0_011[n];
  __syncthreads();

  float U[16], mP = NEG_INF;
  for (int p = 0; p < 16; p++) {
    float a = bA;
    for (int c = 0; c < 4; c++) a += wA[c] * x110[((b * 4 + c) * 16 + p) * 128 + g];
    a += cc[p];
    U[p] = a;
    mP = fmaxf(mP, a);
  }

  const float* xr0 = x011 + (b * 2 + 0) * 12288 + g * 96;
  const float* xr1 = x011 + (b * 2 + 1) * 12288 + g * 96;
  float mE = NEG_INF;
  _Float16* trow = t011h + n * 24576 + b * 12288 + g;
  for (int e = 0; e < 96; e++) {
    float V = bD + wD0 * xr0[e] + wD1 * xr1[e] + gg + ee[e];
    mE = fmaxf(mE, V);
    float s = mP + V;                 // s011 = leaky(maxp U + V)
    trow[e * 128] = (_Float16)LEAKY(s);
  }
  {
    float s = mP + mE;                // maxp t110 == maxe t011 == leaky(mP+mE)
    bfh[n * 256 + b * 128 + g] = (_Float16)LEAKY(s);
  }
  for (int p = 0; p < 16; p++) {
    float s = U[p] + mE;              // s110 = leaky(U + maxe V)
    float v = LEAKY(s);
    t110h[n * 4096 + b * 2048 + p * 128 + g] = (_Float16)v;
    tl[p * 128 + g] = v;
  }
  __syncthreads();
  if (g < 16) {                       // max over g of t110
    float m = NEG_INF;
    for (int q = 0; q < 128; q++) m = fmaxf(m, tl[g * 128 + q]);
    mg110h[n * 32 + b * 16 + g] = (_Float16)m;
  }
  if (g < 96) {                       // s001 = max over g of s011
    float m = NEG_INF;
    const _Float16* tr = t011h + n * 24576 + b * 12288 + g * 128;
    for (int q = 0; q < 128; q++) m = fmaxf(m, (float)tr[q]);
    s001h[n * 192 + b * 96 + g] = (_Float16)m;
  }
}

// ---------------------------------------------------------------------------
// Transpose [128][COLS] f16 -> [COLS][128] f16 so GEMM B-fragments become a
// single contiguous 32B chunk per lane. LDS-tiled: coalesced on both sides.
// Block: 128 threads handles 32 columns x 128 k.
// ---------------------------------------------------------------------------
template <int COLS>
__global__ __launch_bounds__(128) void k_tr(const _Float16* __restrict__ in,
                                            _Float16* __restrict__ out) {
  int colBase = blockIdx.x * 32;
  int t = threadIdx.x;
  __shared__ _Float16 tile[128][34];   // +2 pad to break bank conflicts

  int c = t & 31, k0 = t >> 5;         // 4 k-rows per sweep
#pragma unroll
  for (int i = 0; i < 32; i++) {
    int k = k0 * 1 + i * 4;
    tile[k][c] = in[k * COLS + colBase + c];
  }
  __syncthreads();

  int cc = t >> 2, q = t & 3;          // thread -> (column, k-quarter)
  _Float16* op = out + (colBase + cc) * 128 + q * 32;
#pragma unroll
  for (int j = 0; j < 32; j += 8) {
    v8h v;
#pragma unroll
    for (int m = 0; m < 8; m++) v[m] = tile[q * 32 + j + m][cc];
    *(v8h*)(op + j) = v;
  }
}

// ---------------------------------------------------------------------------
// WMMA GEMM: Out[128][COLS] = Aw[128x128] * Act[128][COLS], with Act given
// transposed as BactT[COLS][128]. One wave per 16x16 tile, K=128 in 4 steps
// of v_wmma_f32_16x16x32_f16. Fragment layouts (wave32); A/B use this code's contiguous k order, D as in the ISA:
//   A elem  m    -> K = m + 16*half + 32*ks
//   B elem  m    -> K = m + 16*half + 32*ks  (contiguous in BactT row!)
//   D elem  r    -> row M = mt*16 + r + 8*half, col = ct*16 + (lane&15)
// ---------------------------------------------------------------------------
template <int COLS>
__global__ __launch_bounds__(128) void k_gemm(
    const _Float16* __restrict__ Aw, const _Float16* __restrict__ BactT,
    float* __restrict__ Out) {
  int wave = threadIdx.x >> 5;
  int lane = threadIdx.x & 31;
  int tile = blockIdx.x * 4 + wave;
  int mt = tile & 7;        // M tile (8 total: 128/16)
  int ct = tile >> 3;       // column tile
  int half = lane >> 4;
  int ln = lane & 15;
  int col = ct * 16 + ln;

  const _Float16* arow = Aw + (mt * 16 + ln) * 128 + 16 * half;
  const _Float16* brow = BactT + col * 128 + 16 * half;

  v8f acc = {};
#pragma unroll
  for (int ks = 0; ks < 4; ks++) {
    v8h alo = *(const v8h*)(arow + ks * 32);
    v8h ahi = *(const v8h*)(arow + ks * 32 + 8);
    v16h a = __builtin_shufflevector(alo, ahi, 0, 1, 2, 3, 4, 5, 6, 7,
                                     8, 9, 10, 11, 12, 13, 14, 15);
    v16h bf = *(const v16h*)(brow + ks * 32);   // 32B contiguous fragment
    acc = __builtin_amdgcn_wmma_f32_16x16x32_f16(false, a, false, bf,
                                                 (short)0, acc, false, false);
  }
  float* op = Out + (mt * 16 + 8 * half) * COLS + col;
#pragma unroll
  for (int r = 0; r < 8; r++) op[r * COLS] = acc[r];
}

// ---------------------------------------------------------------------------
// Layer 1 epilogue: one block per (b,n). Applies biases, does the monotone
// max-collapse and head reductions:
//   H110[b,n,p] = mean_g s110_2, H011e = mean_g s011_2, H001e = max_g s011_2,
//   H011m = mean_{g,e} s011_2, H001m = mean_e s001_2
// ---------------------------------------------------------------------------
__global__ __launch_bounds__(128) void k_r1(
    const float* __restrict__ A1, const float* __restrict__ D1,
    const float* __restrict__ BF1, const float* __restrict__ C1,
    const float* __restrict__ EG1, const float* __restrict__ b1,
    float* __restrict__ H110, float* __restrict__ H011m,
    float* __restrict__ H011e, float* __restrict__ H001e,
    float* __restrict__ H001m) {
  int b = blockIdx.x >> 7, n = blockIdx.x & 127;
  int g = threadIdx.x;
  __shared__ float eg[96], cl[16], mPa[128], bgl[128], tl[16 * 128];
  __shared__ float sume[96], maxe[96];

  float b0 = b1[n], b1v = b1[128 + n], b2 = b1[256 + n], b3 = b1[384 + n];
  float b4 = b1[512 + n], b5 = b1[640 + n], b6 = b1[768 + n];

  if (g < 96) eg[g] = EG1[n * 192 + b * 96 + g] + b4 + b6;
  if (g < 16) cl[g] = C1[n * 32 + b * 16 + g] + b2;
  float bg = BF1[n * 256 + b * 128 + g] + b1v + b5;
  bgl[g] = bg;
  __syncthreads();

  float U[16], mP = NEG_INF;
  const float* ap = A1 + n * 4096 + b * 2048 + g;
  for (int p = 0; p < 16; p++) {
    float u = ap[p * 128] + b0 + cl[p];
    U[p] = u;
    mP = fmaxf(mP, u);
  }
  mPa[g] = mP;

  const float* dp = D1 + n * 24576 + b * 12288 + g;
  float mE = NEG_INF;
  for (int e = 0; e < 96; e++) mE = fmaxf(mE, dp[e * 128] + bg + b3 + eg[e]);

  for (int p = 0; p < 16; p++) {
    float s = U[p] + mE;
    tl[p * 128 + g] = LEAKY(s);
  }
  __syncthreads();

  if (g < 16) {
    float s = 0.0f;
    for (int q = 0; q < 128; q++) s += tl[g * 128 + q];
    H110[(b * 128 + n) * 16 + g] = s * (1.0f / 128.0f);
  }
  if (g < 96) {
    const float* de = D1 + n * 24576 + b * 12288 + g * 128;
    float ege = eg[g];
    float sg = 0.0f, mg = NEG_INF;
    for (int q = 0; q < 128; q++) {
      float s = mPa[q] + de[q] + bgl[q] + b3 + ege;
      float v = LEAKY(s);
      sg += v;
      mg = fmaxf(mg, v);
    }
    H011e[(b * 128 + n) * 96 + g] = sg * (1.0f / 128.0f);
    H001e[(b * 128 + n) * 96 + g] = mg;
    sume[g] = sg;
    maxe[g] = mg;
  }
  __syncthreads();
  if (g == 0) {
    float st = 0.0f, sm = 0.0f;
    for (int e = 0; e < 96; e++) { st += sume[e]; sm += maxe[e]; }
    H011m[b * 128 + n] = st * (1.0f / (128.0f * 96.0f));
    H001m[b * 128 + n] = sm * (1.0f / 96.0f);
  }
}

// ---------------------------------------------------------------------------
// Heads: actions (64 floats) + values (4 floats), single block
// ---------------------------------------------------------------------------
__global__ __launch_bounds__(128) void k_heads(
    const float* __restrict__ H110, const float* __restrict__ H011m,
    const float* __restrict__ H011e, const float* __restrict__ H001e,
    const float* __restrict__ H001m,
    const float* __restrict__ Wact, const float* __restrict__ bact,
    const float* __restrict__ Wcrit, const float* __restrict__ bcrit,
    float* __restrict__ out) {
  int t = threadIdx.x;
  __shared__ float c110s[64], ces[384], vp[64], ve[384];

  if (t < 64) {                       // t = b*32 + co*16 + p
    int b = t >> 5, co = (t >> 4) & 1, p = t & 15;
    float a0 = 0, a1 = 0, a2 = 0, c0 = 0;
    for (int n = 0; n < 128; n++) {
      float h = H110[(b * 128 + n) * 16 + p];
      a0 += Wact[co * 128 + n] * h;
      c0 += Wcrit[co * 128 + n] * h;
      a1 += Wact[(2 + co) * 128 + n] * H011m[b * 128 + n];
      a2 += Wact[(4 + co) * 128 + n] * H001m[b * 128 + n];
    }
    float s = a0 + bact[co] + a1 + bact[2 + co] + a2 + bact[4 + co];
    out[(b * 2 + co) * 16 + p] = LEAKY(s);
    c110s[t] = c0 + bcrit[co];
  }
  for (int idx = t; idx < 384; idx += 128) {   // idx = b*192 + co*96 + e
    int b = idx / 192, r = idx % 192, co = r / 96, e = r % 96;
    float s1 = 0, s2 = 0;
    for (int n = 0; n < 128; n++) {
      s1 += Wcrit[(2 + co) * 128 + n] * H011e[(b * 128 + n) * 96 + e];
      s2 += Wcrit[(4 + co) * 128 + n] * H001e[(b * 128 + n) * 96 + e];
    }
    ces[idx] = s1 + bcrit[2 + co] + s2 + bcrit[4 + co];
  }
  __syncthreads();

  if (t < 64) {
    int b = t >> 5, co = (t >> 4) & 1;
    float m = NEG_INF;
    float c = c110s[t];
    for (int e = 0; e < 96; e++) {
      float v = c + ces[b * 192 + co * 96 + e];
      m = fmaxf(m, LEAKY(v));
    }
    vp[t] = m;
  }
  for (int idx = t; idx < 384; idx += 128) {
    int b = idx / 192, r = idx % 192, co = r / 96;
    float ce = ces[idx];
    float m = NEG_INF;
    for (int p = 0; p < 16; p++) {
      float v = c110s[b * 32 + co * 16 + p] + ce;
      m = fmaxf(m, LEAKY(v));
    }
    ve[idx] = m;
  }
  __syncthreads();
  if (t < 4) {
    int b = t >> 1, co = t & 1;
    float v110 = 0.0f, v011 = 0.0f;
    for (int p = 0; p < 16; p++) v110 += vp[b * 32 + co * 16 + p];
    for (int e = 0; e < 96; e++) v011 += ve[b * 192 + co * 96 + e];
    out[64 + b * 2 + co] = 2.0f + v110 + 2.0f * v011;
  }
}

// ---------------------------------------------------------------------------
extern "C" void kernel_launch(void* const* d_in, const int* in_sizes, int n_in,
                              void* d_out, int out_size, void* d_ws, size_t ws_size,
                              hipStream_t stream) {
  const float* x110   = (const float*)d_in[0];
  const float* x011   = (const float*)d_in[1];
  const float* x001   = (const float*)d_in[2];
  const float* W0_110 = (const float*)d_in[3];
  const float* b0_110 = (const float*)d_in[4];
  const float* W0_011 = (const float*)d_in[5];
  const float* b0_011 = (const float*)d_in[6];
  const float* W0_001 = (const float*)d_in[7];
  const float* b0_001 = (const float*)d_in[8];
  const float* W1     = (const float*)d_in[9];
  const float* b1     = (const float*)d_in[10];
  const float* Wact   = (const float*)d_in[11];
  const float* bact   = (const float*)d_in[12];
  const float* Wcrit  = (const float*)d_in[13];
  const float* bcrit  = (const float*)d_in[14];
  float* out = (float*)d_out;

  char* ws = (char*)d_ws;
  size_t off = 0;
  auto alloc = [&](size_t bytes) -> char* {
    char* p = ws + off;
    off = (off + bytes + 255) & ~(size_t)255;
    return p;
  };
  _Float16* t110h  = (_Float16*)alloc(128 * 4096 * 2);
  _Float16* t011h  = (_Float16*)alloc((size_t)128 * 24576 * 2);
  _Float16* bfh    = (_Float16*)alloc(128 * 256 * 2);
  _Float16* mg110h = (_Float16*)alloc(128 * 32 * 2);
  _Float16* s001h  = (_Float16*)alloc(128 * 192 * 2);
  _Float16* t110T  = (_Float16*)alloc(128 * 4096 * 2);
  _Float16* t011T  = (_Float16*)alloc((size_t)128 * 24576 * 2);
  _Float16* bfT    = (_Float16*)alloc(128 * 256 * 2);
  _Float16* mgT    = (_Float16*)alloc(128 * 32 * 2);
  _Float16* s001T  = (_Float16*)alloc(128 * 192 * 2);
  _Float16* wf     = (_Float16*)alloc(5 * 16384 * 2);
  float* A1  = (float*)alloc(128 * 4096 * 4);
  float* D1  = (float*)alloc((size_t)128 * 24576 * 4);
  float* BF1 = (float*)alloc(128 * 256 * 4);
  float* C1  = (float*)alloc(128 * 32 * 4);
  float* EG1 = (float*)alloc(128 * 192 * 4);
  float* GG0 = (float*)alloc(2 * 128 * 128 * 4);
  float* EE0 = (float*)alloc(2 * 128 * 96 * 4);
  float* CC0 = (float*)alloc(2 * 128 * 16 * 4);
  float* H110  = (float*)alloc(2 * 128 * 16 * 4);
  float* H011m = (float*)alloc(256 * 4);
  float* H011e = (float*)alloc(2 * 128 * 96 * 4);
  float* H001e = (float*)alloc(2 * 128 * 96 * 4);
  float* H001m = (float*)alloc(256 * 4);

  k_wprep<<<320, 256, 0, stream>>>(W1, wf);
  k_l0a<<<2, 128, 0, stream>>>(x110, x011, x001, W0_110, b0_110, W0_011,
                               b0_011, W0_001, b0_001, GG0, EE0, CC0);
  k_l0b<<<256, 128, 0, stream>>>(x110, x011, W0_110, b0_110, W0_011, b0_011,
                                 GG0, EE0, CC0, t110h, t011h, bfh, mg110h, s001h);

  // Transpose activations to [col][k] so B-fragments are lane-contiguous
  k_tr<4096><<<128, 128, 0, stream>>>(t110h, t110T);
  k_tr<24576><<<768, 128, 0, stream>>>(t011h, t011T);
  k_tr<256><<<8, 128, 0, stream>>>(bfh, bfT);
  k_tr<32><<<1, 128, 0, stream>>>(mg110h, mgT);
  k_tr<192><<<6, 128, 0, stream>>>(s001h, s001T);

  // Five WMMA GEMMs: grid = (8 * cols/16) tiles / 4 waves-per-block = cols/8
  k_gemm<4096><<<512, 128, 0, stream>>>(wf + 0 * 16384, t110T, A1);
  k_gemm<24576><<<3072, 128, 0, stream>>>(wf + 1 * 16384, t011T, D1);
  k_gemm<256><<<32, 128, 0, stream>>>(wf + 2 * 16384, bfT, BF1);
  k_gemm<32><<<4, 128, 0, stream>>>(wf + 3 * 16384, mgT, C1);
  k_gemm<192><<<24, 128, 0, stream>>>(wf + 4 * 16384, s001T, EG1);

  k_r1<<<256, 128, 0, stream>>>(A1, D1, BF1, C1, EG1, b1,
                                H110, H011m, H011e, H001e, H001m);
  k_heads<<<1, 128, 0, stream>>>(H110, H011m, H011e, H001e, H001m,
                                 Wact, bact, Wcrit, bcrit, out);
}
